// MultiModalSentimentModel_31731218383431
// MI455X (gfx1250) — matrix operand order repaired and checked by reading, not compiled, not run on hardware
//
#include <hip/hip_runtime.h>
#include <hip/hip_bf16.h>

typedef __attribute__((ext_vector_type(16))) _Float16 half16;
typedef __attribute__((ext_vector_type(8)))  _Float16 half8;
typedef __attribute__((ext_vector_type(8)))  float    float8;

__device__ __forceinline__ half16 cat16(half8 lo, half8 hi) {
  return __builtin_shufflevector(lo, hi, 0,1,2,3,4,5,6,7,8,9,10,11,12,13,14,15);
}

// A fragment (16x32 f16), row handled by caller (row = lane%16).
// elems 0..15 -> K = khalf*16 + 0..15 (contiguous)
__device__ __forceinline__ half16 load_afrag(const _Float16* rowp, int lane) {
  int khalf = (lane >> 4) & 1;
  half8 lo = *(const half8*)(rowp + khalf * 16);
  half8 hi = *(const half8*)(rowp + khalf * 16 + 8);
  return cat16(lo, hi);
}

// B fragment (32x16 f16), col n handled by caller (n = lane%16), data stored [n][k].
// elems 0..15 -> K = khalf*16 + 0..15 (contiguous)
__device__ __forceinline__ half16 load_bfrag(const _Float16* colp, int lane) {
  int khalf = (lane >> 4) & 1;
  half8 lo = *(const half8*)(colp + khalf * 16);
  half8 hi = *(const half8*)(colp + khalf * 16 + 8);
  return cat16(lo, hi);
}

__device__ __forceinline__ float8 wmma_f16(half16 a, half16 b, float8 c) {
  return __builtin_amdgcn_wmma_f32_16x16x32_f16(false, a, false, b, (short)0, c, false, false);
}

__device__ __forceinline__ float fast_sig(float x) {
  return __builtin_amdgcn_rcpf(1.f + __expf(-x));
}
__device__ __forceinline__ float fast_tanh(float x) {
  return 2.f * __builtin_amdgcn_rcpf(1.f + __expf(-2.f * x)) - 1.f;
}

// CDNA5 async memory->LDS copy (ASYNCcnt path), 16 bytes per lane.
__device__ __forceinline__ void async_copy_b128(unsigned lds_off, const void* gaddr) {
  asm volatile("global_load_async_to_lds_b128 %0, %1, off"
               :: "v"(lds_off), "v"((unsigned long long)(size_t)gaddr)
               : "memory");
}
__device__ __forceinline__ void wait_async0() {
  asm volatile("s_wait_asynccnt 0" ::: "memory");
}

// ---------------------------------------------------------------------------
// Generic tiled WMMA GEMM: C[m][n] = act(alpha * sum_k A[m][k]*B(n,k) + bias + resid)
// B_KN=false: B row-major N x K (weights). B_KN=true: B is K x N in memory.
// Block tile 128x128, K-step 32, 256 threads = 8 waves, wave tile 64x32.
// A (and B in NK mode) staged with GLOBAL_LOAD_ASYNC_TO_LDS_B128.
// ---------------------------------------------------------------------------
template<bool B_KN>
__global__ __launch_bounds__(256)
void gemm_wmma_kernel(const _Float16* __restrict__ A, long sA, int lda,
                      const _Float16* __restrict__ B, long sB, int ldb,
                      const float* __restrict__ bias,
                      const float* __restrict__ resid, int ldres, long sRes,
                      float alpha, int relu,
                      float* __restrict__ out32, int ldc32, long sC32,
                      _Float16* __restrict__ out16, int ldc16, long sC16,
                      int K)
{
  __shared__ __align__(16) _Float16 a_sh[128 * 32]; // [m][k]
  __shared__ __align__(16) _Float16 b_sh[128 * 32]; // [n][k]
  const int bz = blockIdx.z;
  A += (long)bz * sA;
  B += (long)bz * sB;
  const int tile_m = blockIdx.y * 128, tile_n = blockIdx.x * 128;
  const int t = threadIdx.x, lane = t & 31, w = t >> 5;
  const int wm = (w >> 2) * 64, wn = (w & 3) * 32;
  const int ln = lane & 15, lh = lane >> 4;
  float8 acc[4][2] = {};
  for (int k0 = 0; k0 < K; k0 += 32) {
    __syncthreads();
    {
      int row = t >> 1, kc = (t & 1) * 16;
      const _Float16* g = A + (long)(tile_m + row) * lda + k0 + kc;
      unsigned lo = (unsigned)(size_t)(a_sh + row * 32 + kc);
      async_copy_b128(lo, g);
      async_copy_b128(lo + 16, g + 8);
    }
    if (!B_KN) {
      int row = t >> 1, kc = (t & 1) * 16;
      const _Float16* g = B + (long)(tile_n + row) * ldb + k0 + kc;
      unsigned lo = (unsigned)(size_t)(b_sh + row * 32 + kc);
      async_copy_b128(lo, g);
      async_copy_b128(lo + 16, g + 8);
    } else {
      int kk = t >> 3, nc = (t & 7) * 16;
      const _Float16* g = B + (long)(k0 + kk) * ldb + tile_n + nc;
      half8 x0 = *(const half8*)g;
      half8 x1 = *(const half8*)(g + 8);
      #pragma unroll
      for (int q = 0; q < 8; q++) b_sh[(nc + q) * 32 + kk] = x0[q];
      #pragma unroll
      for (int q = 0; q < 8; q++) b_sh[(nc + 8 + q) * 32 + kk] = x1[q];
    }
    wait_async0();
    __syncthreads();
    half16 bf0 = load_bfrag(b_sh + (wn + ln) * 32, lane);
    half16 bf1 = load_bfrag(b_sh + (wn + 16 + ln) * 32, lane);
    #pragma unroll
    for (int mf = 0; mf < 4; mf++) {
      half16 af = load_afrag(a_sh + (wm + mf * 16 + ln) * 32, lane);
      acc[mf][0] = wmma_f16(af, bf0, acc[mf][0]);
      acc[mf][1] = wmma_f16(af, bf1, acc[mf][1]);
    }
  }
  #pragma unroll
  for (int mf = 0; mf < 4; mf++)
    #pragma unroll
    for (int nf = 0; nf < 2; nf++)
      #pragma unroll
      for (int r = 0; r < 8; r++) {
        int row = tile_m + wm + mf * 16 + r + lh * 8;
        int col = tile_n + wn + nf * 16 + ln;
        float v = acc[mf][nf][r] * alpha;
        if (bias)  v += bias[col];
        if (resid) v += resid[(long)bz * sRes + (long)row * ldres + col];
        if (relu)  v = fmaxf(v, 0.f);
        if (out32) out32[(long)bz * sC32 + (long)row * ldc32 + col] = v;
        if (out16) out16[(long)bz * sC16 + (long)row * ldc16 + col] = (_Float16)v;
      }
}

// ---------------------------------------------------------------------------
// Persistent bidirectional-LSTM recurrence. 4 blocks: (acoustic|visual) x (fwd|rev).
// 512 threads = 16 waves, each wave owns 6 N-tiles (96 total), processed as
// 3 passes of 2 tiles so only 2 accumulators (16 VGPRs) are live in the K loop.
// h (16x384 f16, rows 8..15 zero), c (8x384 f32), gates (8x1536 f32) in LDS.
// ---------------------------------------------------------------------------
__global__ __launch_bounds__(512)
void bilstm_kernel(const _Float16* __restrict__ xs0, const _Float16* __restrict__ xs1,
                   const _Float16* __restrict__ xs2, const _Float16* __restrict__ xs3,
                   const _Float16* __restrict__ uf0, const _Float16* __restrict__ uf1,
                   const _Float16* __restrict__ uf2, const _Float16* __restrict__ uf3,
                   const float* __restrict__ bh0, const float* __restrict__ bh1,
                   const float* __restrict__ bh2, const float* __restrict__ bh3,
                   _Float16* __restrict__ out /* fused_in, ld 2304 */)
{
  __shared__ __align__(16) _Float16 h_sh[16 * 384]; // [m][k]
  __shared__ float gates_sh[8 * 1536];
  __shared__ float c_sh[8 * 384];
  const int blk = blockIdx.x;
  const _Float16* xs = blk == 0 ? xs0 : blk == 1 ? xs1 : blk == 2 ? xs2 : xs3;
  const _Float16* uf = blk == 0 ? uf0 : blk == 1 ? uf1 : blk == 2 ? uf2 : uf3;
  const float*   bhh = blk == 0 ? bh0 : blk == 1 ? bh1 : blk == 2 ? bh2 : bh3;
  const int col = 768 + blk * 384;
  const int rev = blk & 1;
  const int t = threadIdx.x, lane = t & 31, w = t >> 5;
  const int ln = lane & 15, lh = lane >> 4;
  for (int i = t; i < 16 * 384; i += 512) h_sh[i] = (_Float16)0.f;
  for (int i = t; i < 8 * 384; i += 512) c_sh[i] = 0.f;
  __syncthreads();
  const int nt0 = w * 6;
  for (int s = 0; s < 512; s++) {
    const int tt = rev ? 511 - s : s;
    #pragma unroll 1
    for (int pr = 0; pr < 3; pr++) {
      const int ntA = nt0 + pr * 2;
      const int nA = ntA * 16 + ln;
      const int nB = nA + 16;
      float8 accA = {}, accB = {};
      if (lh == 0) {
        const _Float16* xrow = xs + (long)tt * 1536;
        float bA = bhh[nA], bB = bhh[nB];
        #pragma unroll
        for (int r = 0; r < 8; r++) {
          accA[r] = (float)xrow[(long)r * 512 * 1536 + nA] + bA;
          accB[r] = (float)xrow[(long)r * 512 * 1536 + nB] + bB;
        }
      }
      const _Float16* ufA = uf + ((long)ntA * 12 * 32 + lane) * 16;
      #pragma unroll 2
      for (int ks = 0; ks < 12; ks++) {
        half16 af = load_afrag(h_sh + ln * 384 + ks * 32, lane);
        half16 bfA = *(const half16*)(ufA + (long)ks * 32 * 16);
        half16 bfB = *(const half16*)(ufA + ((long)(12 + ks)) * 32 * 16);
        accA = wmma_f16(af, bfA, accA);
        accB = wmma_f16(af, bfB, accB);
      }
      if (lh == 0) {
        #pragma unroll
        for (int r = 0; r < 8; r++) {
          gates_sh[r * 1536 + nA] = accA[r];
          gates_sh[r * 1536 + nB] = accB[r];
        }
      }
    }
    __syncthreads();
    for (int i = t; i < 8 * 384; i += 512) {
      int b = i / 384, j = i - b * 384;
      float gi = gates_sh[b * 1536 + j];
      float gf = gates_sh[b * 1536 + 384 + j];
      float gg = gates_sh[b * 1536 + 768 + j];
      float go = gates_sh[b * 1536 + 1152 + j];
      float c = fast_sig(gf) * c_sh[i] + fast_sig(gi) * fast_tanh(gg);
      c_sh[i] = c;
      float hh = fast_sig(go) * fast_tanh(c);
      h_sh[b * 384 + j] = (_Float16)hh;
      out[((long)b * 512 + tt) * 2304 + col + j] = (_Float16)hh;
    }
    __syncthreads();
  }
}

// ---------------------------------------------------------------------------
// Flash-style multi-head self-attention for encoder (12 heads, d=64).
// grid (qtile=4, head=12, batch=8), 8 waves x 16 q-rows, online softmax.
// ---------------------------------------------------------------------------
__global__ __launch_bounds__(256)
void attn_encoder_kernel(const _Float16* __restrict__ qkv, _Float16* __restrict__ o16)
{
  __shared__ __align__(16) _Float16 k_sh[32 * 64];   // [key][d]
  __shared__ __align__(16) _Float16 vT_sh[64 * 32];  // [d][key]
  __shared__ __align__(16) _Float16 p_sh[8][16 * 32];
  const int t = threadIdx.x, lane = t & 31, w = t >> 5;
  const int ln = lane & 15, lh = lane >> 4;
  const int b = blockIdx.z, h = blockIdx.y, q0 = blockIdx.x * 128;
  const _Float16* qbase = qkv + (long)b * 512 * 2304 + h * 64;
  const _Float16* kbase = qbase + 768;
  const _Float16* vbase = qbase + 1536;
  const int qrow = q0 + w * 16 + ln;
  half16 qf0 = load_afrag(qbase + (long)qrow * 2304, lane);
  half16 qf1 = load_afrag(qbase + (long)qrow * 2304 + 32, lane);
  float8 o[4] = {};
  float mrow[8], lrow[8];
  #pragma unroll
  for (int r = 0; r < 8; r++) { mrow[r] = -1e30f; lrow[r] = 0.f; }
  for (int kt = 0; kt < 512; kt += 32) {
    __syncthreads();
    {
      int key = t >> 3, dc = (t & 7) * 8;
      half8 kk = *(const half8*)(kbase + (long)(kt + key) * 2304 + dc);
      *(half8*)(k_sh + key * 64 + dc) = kk;
      half8 vv = *(const half8*)(vbase + (long)(kt + key) * 2304 + dc);
      #pragma unroll
      for (int q = 0; q < 8; q++) vT_sh[(dc + q) * 32 + key] = vv[q];
    }
    __syncthreads();
    float8 s0 = {}, s1 = {};
    s0 = wmma_f16(qf0, load_bfrag(k_sh + ln * 64, lane), s0);
    s0 = wmma_f16(qf1, load_bfrag(k_sh + ln * 64 + 32, lane), s0);
    s1 = wmma_f16(qf0, load_bfrag(k_sh + (ln + 16) * 64, lane), s1);
    s1 = wmma_f16(qf1, load_bfrag(k_sh + (ln + 16) * 64 + 32, lane), s1);
    float p0[8], p1[8], tm[8];
    #pragma unroll
    for (int r = 0; r < 8; r++) {
      p0[r] = s0[r] * 0.125f; p1[r] = s1[r] * 0.125f;
      tm[r] = fmaxf(p0[r], p1[r]);
    }
    #pragma unroll
    for (int off = 1; off < 16; off <<= 1)
      #pragma unroll
      for (int r = 0; r < 8; r++) tm[r] = fmaxf(tm[r], __shfl_xor(tm[r], off, 32));
    float al[8], ps[8];
    #pragma unroll
    for (int r = 0; r < 8; r++) {
      float mn = fmaxf(mrow[r], tm[r]);
      al[r] = __expf(mrow[r] - mn);
      mrow[r] = mn;
      p0[r] = __expf(p0[r] - mn);
      p1[r] = __expf(p1[r] - mn);
      p_sh[w][(r + 8 * lh) * 32 + ln]      = (_Float16)p0[r];
      p_sh[w][(r + 8 * lh) * 32 + ln + 16] = (_Float16)p1[r];
      ps[r] = p0[r] + p1[r];
    }
    #pragma unroll
    for (int off = 1; off < 16; off <<= 1)
      #pragma unroll
      for (int r = 0; r < 8; r++) ps[r] += __shfl_xor(ps[r], off, 32);
    #pragma unroll
    for (int r = 0; r < 8; r++) lrow[r] = lrow[r] * al[r] + ps[r];
    #pragma unroll
    for (int nf = 0; nf < 4; nf++)
      #pragma unroll
      for (int r = 0; r < 8; r++) o[nf][r] *= al[r];
    asm volatile("s_wait_dscnt 0" ::: "memory"); // wave-private LDS RAW (P staging)
    half16 pf = load_afrag(&p_sh[w][ln * 32], lane);
    #pragma unroll
    for (int nf = 0; nf < 4; nf++) {
      half16 bv = load_bfrag(vT_sh + (nf * 16 + ln) * 32, lane);
      o[nf] = wmma_f16(pf, bv, o[nf]);
    }
  }
  float inv[8];
  #pragma unroll
  for (int r = 0; r < 8; r++) inv[r] = __builtin_amdgcn_rcpf(lrow[r]);
  #pragma unroll
  for (int nf = 0; nf < 4; nf++)
    #pragma unroll
    for (int r = 0; r < 8; r++) {
      int row = b * 512 + q0 + w * 16 + r + 8 * lh;
      o16[(long)row * 768 + h * 64 + nf * 16 + ln] = (_Float16)(o[nf][r] * inv[r]);
    }
}

// ---------------------------------------------------------------------------
__global__ __launch_bounds__(256)
void softmax512_kernel(const float* __restrict__ in, _Float16* __restrict__ out)
{
  __shared__ float red[256];
  const long row = blockIdx.x;
  const int t = threadIdx.x;
  const float* p = in + row * 512;
  float v0 = p[t], v1 = p[t + 256];
  red[t] = fmaxf(v0, v1);
  __syncthreads();
  for (int k = 128; k > 0; k >>= 1) { if (t < k) red[t] = fmaxf(red[t], red[t + k]); __syncthreads(); }
  float m = red[0];
  __syncthreads();
  float e0 = __expf(v0 - m), e1 = __expf(v1 - m);
  red[t] = e0 + e1;
  __syncthreads();
  for (int k = 128; k > 0; k >>= 1) { if (t < k) red[t] += red[t + k]; __syncthreads(); }
  float inv = 1.f / red[0];
  out[row * 512 + t]       = (_Float16)(e0 * inv);
  out[row * 512 + t + 256] = (_Float16)(e1 * inv);
}

__global__ __launch_bounds__(256)
void layernorm_kernel(const float* __restrict__ in, const float* __restrict__ g,
                      const float* __restrict__ be,
                      float* __restrict__ out32, _Float16* __restrict__ out16)
{
  __shared__ float red[256];
  const long row = blockIdx.x;
  const int t = threadIdx.x;
  const float* p = in + row * 768;
  float x0 = p[t], x1 = p[t + 256], x2 = p[t + 512];
  red[t] = x0 + x1 + x2;
  __syncthreads();
  for (int k = 128; k > 0; k >>= 1) { if (t < k) red[t] += red[t + k]; __syncthreads(); }
  float mean = red[0] * (1.f / 768.f);
  __syncthreads();
  float d0 = x0 - mean, d1 = x1 - mean, d2 = x2 - mean;
  red[t] = d0 * d0 + d1 * d1 + d2 * d2;
  __syncthreads();
  for (int k = 128; k > 0; k >>= 1) { if (t < k) red[t] += red[t + k]; __syncthreads(); }
  float rstd = rsqrtf(red[0] * (1.f / 768.f) + 1e-5f);
  float y0 = d0 * rstd * g[t] + be[t];
  float y1 = d1 * rstd * g[t + 256] + be[t + 256];
  float y2 = d2 * rstd * g[t + 512] + be[t + 512];
  if (out32) { out32[row * 768 + t] = y0; out32[row * 768 + t + 256] = y1; out32[row * 768 + t + 512] = y2; }
  if (out16) { out16[row * 768 + t] = (_Float16)y0; out16[row * 768 + t + 256] = (_Float16)y1; out16[row * 768 + t + 512] = (_Float16)y2; }
}

__global__ __launch_bounds__(256)
void f32_to_f16_kernel(const float* __restrict__ in, _Float16* __restrict__ out, long n)
{
  long i = (long)blockIdx.x * 256 + threadIdx.x;
  if (i < n) out[i] = (_Float16)in[i];
}

// Pack Whh (1536 x 384, gates x hidden) into WMMA B-fragment order:
// [n_tile(96)][k_step(12)][lane(32)][elem(16)] where value = Whh[n][k].
__global__ __launch_bounds__(256)
void pack_whh_kernel(const float* __restrict__ U, _Float16* __restrict__ out)
{
  int idx = blockIdx.x * 256 + threadIdx.x;
  if (idx >= 96 * 12 * 32 * 16) return;
  int e = idx & 15;
  int lane = (idx >> 4) & 31;
  int ks = (idx >> 9) % 12;
  int nt = idx / 6144;
  int n = nt * 16 + (lane & 15);
  int k = ks * 32 + (lane >> 4) * 16 + e;
  out[idx] = (_Float16)U[(long)n * 384 + k];
}

__global__ __launch_bounds__(256)
void clf_final_kernel(const _Float16* __restrict__ hid, const float* __restrict__ w2,
                      const float* __restrict__ b2, float* __restrict__ out)
{
  __shared__ float red[256];
  const int row = blockIdx.x, t = threadIdx.x;
  float s = 0.f;
  for (int i = t; i < 768; i += 256) s += (float)hid[(long)row * 768 + i] * w2[i];
  red[t] = s;
  __syncthreads();
  for (int k = 128; k > 0; k >>= 1) { if (t < k) red[t] += red[t + k]; __syncthreads(); }
  if (t == 0) out[row] = red[0] + b2[0];
}

// ---------------------------------------------------------------------------
extern "C" void kernel_launch(void* const* d_in, const int* in_sizes, int n_in,
                              void* d_out, int out_size, void* d_ws, size_t ws_size,
                              hipStream_t stream)
{
  (void)in_sizes; (void)n_in; (void)out_size; (void)ws_size;
  int ii = 0;
  const float* linguistic = (const float*)d_in[ii++];
  const float* acoustic   = (const float*)d_in[ii++];
  const float* visual     = (const float*)d_in[ii++];
  const float* lp_W = (const float*)d_in[ii++];
  const float* lp_b = (const float*)d_in[ii++];
  const float* aWf  = (const float*)d_in[ii++]; const float* aUf  = (const float*)d_in[ii++];
  const float* abif = (const float*)d_in[ii++]; const float* abhf = (const float*)d_in[ii++];
  const float* aWr  = (const float*)d_in[ii++]; const float* aUr  = (const float*)d_in[ii++];
  const float* abir = (const float*)d_in[ii++]; const float* abhr = (const float*)d_in[ii++];
  const float* vWf  = (const float*)d_in[ii++]; const float* vUf  = (const float*)d_in[ii++];
  const float* vbif = (const float*)d_in[ii++]; const float* vbhf = (const float*)d_in[ii++];
  const float* vWr  = (const float*)d_in[ii++]; const float* vUr  = (const float*)d_in[ii++];
  const float* vbir = (const float*)d_in[ii++]; const float* vbhr = (const float*)d_in[ii++];
  const float* uff_W = (const float*)d_in[ii++];
  const float* uff_b = (const float*)d_in[ii++];
  const float* Win  = (const float*)d_in[ii++]; const float* binb = (const float*)d_in[ii++];
  const float* Wout = (const float*)d_in[ii++]; const float* bout = (const float*)d_in[ii++];
  const float* g1   = (const float*)d_in[ii++]; const float* be1  = (const float*)d_in[ii++];
  const float* g2   = (const float*)d_in[ii++]; const float* be2  = (const float*)d_in[ii++];
  const float* W1   = (const float*)d_in[ii++]; const float* b1   = (const float*)d_in[ii++];
  const float* W2   = (const float*)d_in[ii++]; const float* b2   = (const float*)d_in[ii++];
  const float* cW1[4]; const float* cb1[4]; const float* cW2[4]; const float* cb2[4];
  for (int c = 0; c < 4; c++) {
    cW1[c] = (const float*)d_in[ii++]; cb1[c] = (const float*)d_in[ii++];
    cW2[c] = (const float*)d_in[ii++]; cb2[c] = (const float*)d_in[ii++];
  }

  char* base = (char*)d_ws;
  size_t off = 0;
  auto alloc = [&](size_t bytes) -> char* {
    char* p = base + off;
    off += (bytes + 255) & ~(size_t)255;
    return p;
  };
  const long M = 4096; // B*S
  _Float16* lin16  = (_Float16*)alloc(M * 128 * 2);
  _Float16* ac16   = (_Float16*)alloc(M * 128 * 2);
  _Float16* vis16  = (_Float16*)alloc(M * 128 * 2);
  _Float16* lpW16  = (_Float16*)alloc(768 * 128 * 2);
  _Float16* aWf16  = (_Float16*)alloc(1536 * 128 * 2);
  _Float16* aWr16  = (_Float16*)alloc(1536 * 128 * 2);
  _Float16* vWf16  = (_Float16*)alloc(1536 * 128 * 2);
  _Float16* vWr16  = (_Float16*)alloc(1536 * 128 * 2);
  _Float16* uffW16 = (_Float16*)alloc(768L * 2304 * 2);
  _Float16* Win16  = (_Float16*)alloc(2304L * 768 * 2);
  _Float16* Wout16 = (_Float16*)alloc(768L * 768 * 2);
  _Float16* W1_16  = (_Float16*)alloc(3072L * 768 * 2);
  _Float16* W2_16  = (_Float16*)alloc(768L * 3072 * 2);
  _Float16* cW16[4];
  for (int c = 0; c < 4; c++) cW16[c] = (_Float16*)alloc(768L * 768 * 2);
  _Float16* ufA_f = (_Float16*)alloc(589824 * 2);
  _Float16* ufA_r = (_Float16*)alloc(589824 * 2);
  _Float16* ufV_f = (_Float16*)alloc(589824 * 2);
  _Float16* ufV_r = (_Float16*)alloc(589824 * 2);
  float*    l_f32    = (float*)alloc(M * 768 * 4);
  _Float16* fused_in = (_Float16*)alloc(M * 2304 * 2); // cols: l | a | v
  _Float16* xs_af = (_Float16*)alloc(M * 1536 * 2);
  _Float16* xs_ar = (_Float16*)alloc(M * 1536 * 2);
  _Float16* xs_vf = (_Float16*)alloc(M * 1536 * 2);
  _Float16* xs_vr = (_Float16*)alloc(M * 1536 * 2);
  _Float16* qkv16 = (_Float16*)alloc(M * 2304 * 2);
  _Float16* o16   = (_Float16*)alloc(M * 768 * 2);
  float*    x1f   = (float*)alloc(M * 768 * 4);
  float*    xlnf  = (float*)alloc(M * 768 * 4);
  _Float16* x16   = (_Float16*)alloc(M * 768 * 2);
  _Float16* f1_16 = (_Float16*)alloc(M * 3072 * 2);
  float*    f2f   = (float*)alloc(M * 768 * 4);
  _Float16* lg16  = (_Float16*)alloc(M * 768 * 2);
  float*    scores  = (float*)alloc(8L * 512 * 512 * 4);
  _Float16* probs16 = (_Float16*)alloc(8L * 512 * 512 * 2);
  _Float16* ag16    = (_Float16*)alloc(M * 768 * 2);
  _Float16* vg16    = (_Float16*)alloc(M * 768 * 2);
  _Float16* fused16 = (_Float16*)alloc(M * 768 * 2);
  _Float16* hid16   = (_Float16*)alloc(M * 768 * 2);

  auto conv = [&](const float* src, _Float16* dst, long n) {
    f32_to_f16_kernel<<<dim3((unsigned)((n + 255) / 256)), dim3(256), 0, stream>>>(src, dst, n);
  };
  auto gemm = [&](bool bkn, const _Float16* A, long sA, int lda,
                  const _Float16* B, long sB, int ldb,
                  const float* bias, const float* res, int ldres, long sRes,
                  float alpha, int relu,
                  float* o32, int ldc32, long sC32,
                  _Float16* oh, int ldc16, long sC16,
                  int Mm, int Nn, int Kk, int batches) {
    dim3 g(Nn / 128, Mm / 128, batches), blk(256);
    if (bkn)
      gemm_wmma_kernel<true><<<g, blk, 0, stream>>>(A, sA, lda, B, sB, ldb, bias, res, ldres, sRes,
                                                    alpha, relu, o32, ldc32, sC32, oh, ldc16, sC16, Kk);
    else
      gemm_wmma_kernel<false><<<g, blk, 0, stream>>>(A, sA, lda, B, sB, ldb, bias, res, ldres, sRes,
                                                     alpha, relu, o32, ldc32, sC32, oh, ldc16, sC16, Kk);
  };

  // 1) Precision staging: f32 -> f16 copies of activations and weights.
  conv(linguistic, lin16, M * 128);
  conv(acoustic,   ac16,  M * 128);
  conv(visual,     vis16, M * 128);
  conv(lp_W, lpW16, 768 * 128);
  conv(aWf, aWf16, 1536 * 128);  conv(aWr, aWr16, 1536 * 128);
  conv(vWf, vWf16, 1536 * 128);  conv(vWr, vWr16, 1536 * 128);
  conv(uff_W, uffW16, 768L * 2304);
  conv(Win,  Win16,  2304L * 768);
  conv(Wout, Wout16, 768L * 768);
  conv(W1, W1_16, 3072L * 768);
  conv(W2, W2_16, 768L * 3072);
  for (int c = 0; c < 4; c++) conv(cW1[c], cW16[c], 768L * 768);
  {
    dim3 g((96 * 12 * 32 * 16 + 255) / 256), blk(256);
    pack_whh_kernel<<<g, blk, 0, stream>>>(aUf, ufA_f);
    pack_whh_kernel<<<g, blk, 0, stream>>>(aUr, ufA_r);
    pack_whh_kernel<<<g, blk, 0, stream>>>(vUf, ufV_f);
    pack_whh_kernel<<<g, blk, 0, stream>>>(vUr, ufV_r);
  }

  // 2) l = linguistic @ lp_W^T + lp_b  -> l_f32 and f16 into fused_in col 0.
  gemm(false, lin16, 0, 128, lpW16, 0, 128, lp_b, nullptr, 0, 0, 1.f, 0,
       l_f32, 768, 0, fused_in, 2304, 0, 4096, 768, 128, 1);

  // 3) LSTM input projections (xs = x @ Wih^T + bih), f16.
  gemm(false, ac16,  0, 128, aWf16, 0, 128, abif, nullptr, 0, 0, 1.f, 0,
       nullptr, 0, 0, xs_af, 1536, 0, 4096, 1536, 128, 1);
  gemm(false, ac16,  0, 128, aWr16, 0, 128, abir, nullptr, 0, 0, 1.f, 0,
       nullptr, 0, 0, xs_ar, 1536, 0, 4096, 1536, 128, 1);
  gemm(false, vis16, 0, 128, vWf16, 0, 128, vbif, nullptr, 0, 0, 1.f, 0,
       nullptr, 0, 0, xs_vf, 1536, 0, 4096, 1536, 128, 1);
  gemm(false, vis16, 0, 128, vWr16, 0, 128, vbir, nullptr, 0, 0, 1.f, 0,
       nullptr, 0, 0, xs_vr, 1536, 0, 4096, 1536, 128, 1);

  // 4) Persistent LSTM recurrence: fills fused_in cols 768..2303 (a then v, fwd|rev).
  bilstm_kernel<<<dim3(4), dim3(512), 0, stream>>>(
      xs_af, xs_ar, xs_vf, xs_vr, ufA_f, ufA_r, ufV_f, ufV_r,
      abhf, abhr, vbhf, vbhr, fused_in);

  // 5) fused = concat(l,a,v) @ uff_W^T + uff_b  (f16 for classifier).
  gemm(false, fused_in, 0, 2304, uffW16, 0, 2304, uff_b, nullptr, 0, 0, 1.f, 0,
       nullptr, 0, 0, fused16, 768, 0, 4096, 768, 2304, 1);

  // 6) Encoder: qkv projection, flash attention, out-proj(+resid), LN1, FFN, LN2.
  gemm(false, fused_in, 0, 2304, Win16, 0, 768, binb, nullptr, 0, 0, 1.f, 0,
       nullptr, 0, 0, qkv16, 2304, 0, 4096, 2304, 768, 1);
  attn_encoder_kernel<<<dim3(4, 12, 8), dim3(256), 0, stream>>>(qkv16, o16);
  gemm(false, o16, 0, 768, Wout16, 0, 768, bout, l_f32, 768, 0, 1.f, 0,
       x1f, 768, 0, nullptr, 0, 0, 4096, 768, 768, 1);
  layernorm_kernel<<<dim3(4096), dim3(256), 0, stream>>>(x1f, g1, be1, xlnf, x16);
  gemm(false, x16, 0, 768, W1_16, 0, 768, b1, nullptr, 0, 0, 1.f, 1,
       nullptr, 0, 0, f1_16, 3072, 0, 4096, 3072, 768, 1);
  gemm(false, f1_16, 0, 3072, W2_16, 0, 3072, b2, xlnf, 768, 0, 1.f, 0,
       f2f, 768, 0, nullptr, 0, 0, 4096, 768, 3072, 1);
  layernorm_kernel<<<dim3(4096), dim3(256), 0, stream>>>(f2f, g2, be2, nullptr, lg16);

  // 7) Guided attention (q = l, kv = a then v): batched score GEMM, softmax, P@kv.
  const float isq = 1.f / sqrtf(768.f);
  for (int mdl = 0; mdl < 2; mdl++) {
    const _Float16* kv = fused_in + 768 + mdl * 768;
    _Float16* dst = mdl == 0 ? ag16 : vg16;
    gemm(false, fused_in, 512L * 2304, 2304, kv, 512L * 2304, 2304,
         nullptr, nullptr, 0, 0, isq, 0,
         scores, 512, 512L * 512, nullptr, 0, 0, 512, 512, 768, 8);
    softmax512_kernel<<<dim3(4096), dim3(256), 0, stream>>>(scores, probs16);
    gemm(true, probs16, 512L * 512, 512, kv, 512L * 2304, 2304,
         nullptr, nullptr, 0, 0, 1.f, 0,
         nullptr, 0, 0, dst, 768, 512L * 768, 512, 768, 512, 8);
  }

  // 8) Classifiers: relu(X @ W1^T + b1) @ W2^T + b2 -> d_out (m, l, a, v).
  float* outp = (float*)d_out;
  const _Float16* cx[4] = { fused16, lg16, ag16, vg16 };
  for (int c = 0; c < 4; c++) {
    gemm(false, cx[c], 0, 768, cW16[c], 0, 768, cb1[c], nullptr, 0, 0, 1.f, 1,
         nullptr, 0, 0, hid16, 768, 0, 4096, 768, 768, 1);
    clf_final_kernel<<<dim3(4096), dim3(256), 0, stream>>>(hid16, cW2[c], cb2[c], outp + c * 4096);
  }
}
